// TransformerEncoderLayer_86809878986793
// MI455X (gfx1250) — matrix operand order repaired and checked by reading, not compiled, not run on hardware
//
#include <hip/hip_runtime.h>
#include <math.h>

// ---- problem constants (match reference) ----
#define B_  2
#define S_  2048
#define D_  768
#define H_  12
#define I_  3072
#define HD_ 64
#define M_  (B_ * S_)          // 4096 rows
#define SCALE_ 0.125f          // 1/sqrt(HD)

typedef __attribute__((ext_vector_type(8)))  _Float16 v8h;
typedef __attribute__((ext_vector_type(16))) _Float16 v16h;
typedef __attribute__((ext_vector_type(8)))  float    v8f;

// ---------------------------------------------------------------------------
// WMMA helpers (CDNA5: wave32, D = A(16x32 f16) * B(32x16 f16) + C(16x16 f32))
// Fragments use contiguous k order (k = 16*(lane/16) + s) for both A and B.
// ---------------------------------------------------------------------------
static __device__ __forceinline__ v8f wmma16(v16h a, v16h b, v8f c) {
  return __builtin_amdgcn_wmma_f32_16x16x32_f16(false, a, false, b, (short)0, c,
                                                false, false);
}

// A fragment: lane m = lane&15 holds row m.  kb = (lane<16)?0:16.
// halves 0..15 = row[kb .. kb+15]
static __device__ __forceinline__ v16h frag_a(const _Float16* row, int kb) {
  v8h lo = *(const v8h*)(row + kb);
  v8h hi = *(const v8h*)(row + kb + 8);
  v16h r;
#pragma unroll
  for (int i = 0; i < 8; ++i) { r[i] = lo[i]; r[i + 8] = hi[i]; }
  return r;
}

// B fragment: lane n = lane&15 holds column n; 16 contiguous K values
// starting at kb2 = (lane<16)?0:16 of a K-major column (i.e. row of B^T).
static __device__ __forceinline__ v16h frag_b(const _Float16* p) {
  v8h lo = *(const v8h*)(p);
  v8h hi = *(const v8h*)(p + 8);
  v16h r;
#pragma unroll
  for (int i = 0; i < 8; ++i) { r[i] = lo[i]; r[i + 8] = hi[i]; }
  return r;
}

static __device__ __forceinline__ float gelu_exact(float v) {
  return 0.5f * v * (1.0f + erff(v * 0.70710678118654752f));
}

// Low 32 bits of a flat shared-aperture address are the LDS byte offset
// (CDNA5 ISA).
static __device__ __forceinline__ unsigned lds_offset(const void* p) {
  return (unsigned)(unsigned long long)(uintptr_t)p;
}

// CDNA5 async Global->LDS 16B transfer (tracked by ASYNCcnt).
static __device__ __forceinline__ void async_load_b128(unsigned lds_off,
                                                       const void* gaddr) {
  asm volatile("global_load_async_to_lds_b128 %0, %1, off"
               :: "v"(lds_off), "v"(gaddr) : "memory");
}
static __device__ __forceinline__ void wait_asynccnt0() {
  asm volatile("s_wait_asynccnt 0" ::: "memory");
}

// ---------------------------------------------------------------------------
// Convert fp32 weight [K][N] -> transposed f16 [N][K] (batched over blockIdx.y)
// ---------------------------------------------------------------------------
__global__ void cvt_transpose(const float* __restrict__ in,
                              _Float16* __restrict__ out, int K, int N) {
  long z = blockIdx.y;
  in  += z * (long)K * N;
  out += z * (long)K * N;
  long total = (long)K * N;
  for (long i = blockIdx.x * (long)blockDim.x + threadIdx.x; i < total;
       i += (long)gridDim.x * blockDim.x) {
    long k = i / N, n = i % N;
    out[n * (long)K + k] = (_Float16)in[i];
  }
}

// ---------------------------------------------------------------------------
// LayerNorm over last dim (Dd). One block per row. Optional f32 / f16 outputs.
// ---------------------------------------------------------------------------
__global__ void ln_kernel(const float* __restrict__ in, const float* __restrict__ g,
                          const float* __restrict__ bia, float* __restrict__ outF,
                          _Float16* __restrict__ outH, int Dd) {
  long row = blockIdx.x;
  const float* xr = in + row * Dd;
  __shared__ float red[64];
  float s = 0.f, s2 = 0.f;
  for (int i = threadIdx.x; i < Dd; i += blockDim.x) {
    float v = xr[i];
    s += v; s2 += v * v;
  }
#pragma unroll
  for (int o = 16; o > 0; o >>= 1) {
    s  += __shfl_xor(s,  o, 32);
    s2 += __shfl_xor(s2, o, 32);
  }
  int wave = threadIdx.x >> 5, lane = threadIdx.x & 31;
  if (lane == 0) { red[wave] = s; red[32 + wave] = s2; }
  __syncthreads();
  if (threadIdx.x == 0) {
    float a = 0.f, a2 = 0.f;
    int nw = blockDim.x >> 5;
    for (int i = 0; i < nw; ++i) { a += red[i]; a2 += red[32 + i]; }
    red[0] = a; red[32] = a2;
  }
  __syncthreads();
  float mean = red[0] / (float)Dd;
  float var  = red[32] / (float)Dd - mean * mean;
  float inv  = rsqrtf(var + 1e-5f);
  for (int i = threadIdx.x; i < Dd; i += blockDim.x) {
    float v = (xr[i] - mean) * inv * g[i] + bia[i];
    if (outF) outF[row * Dd + i] = v;
    if (outH) outH[row * Dd + i] = (_Float16)v;
  }
}

// ---------------------------------------------------------------------------
// Generic WMMA GEMM: out[M,N] = A[M,K](f16) * BT[N,K](f16)^T + bias[N]
//   + optional residual (f32), optional exact GELU, f32 and/or f16 output,
//   optional transposed f16 output ([N][M], used for V).
// Block = 256 threads (8 waves); each wave owns a 16x64 output tile,
// block covers 128(M) x 64(N). Batched over blockIdx.z.
// The shared 64-col B tile (4KB per 32-K step) is staged into LDS once per
// block via async Global->LDS b128 transfers (256 threads x 16B), then all
// 8 waves read B fragments with ds_load -- 8x less global traffic than
// per-wave fragment loads.
// ---------------------------------------------------------------------------
__global__ void gemm_wmma(const _Float16* __restrict__ A,
                          const _Float16* __restrict__ BT,
                          const float* __restrict__ bias,
                          const float* __restrict__ res,
                          float* __restrict__ outF,
                          _Float16* __restrict__ outH,
                          int M, int N, int K,
                          long bt_batch, long bias_batch, long out_batch,
                          int act, int transH) {
  __shared__ _Float16 btile[64 * 32];   // [n:64][k:32] f16 = 4KB
  long z = blockIdx.z;
  BT   += z * bt_batch;
  bias += z * bias_batch;
  if (outH) outH += z * out_batch;
  if (outF) outF += z * out_batch;

  int tid  = threadIdx.x;
  int lane = tid & 31, wave = tid >> 5;
  int row0 = blockIdx.y * 128 + wave * 16;   // grids are exact: row0 < M always
  int col0 = blockIdx.x * 64;

  int n   = lane & 15;
  int kbA = (lane < 16) ? 0 : 16;
  int kbB = (lane < 16) ? 0 : 16;

  const _Float16* arow = A + (long)(row0 + n) * K;   // lane m == lane&15

  // async-staging assignment: thread tid moves 8 halves (16B) of row tid/4
  int brow = tid >> 2, bquad = tid & 3;
  const _Float16* bsrc = BT + (long)(col0 + brow) * K + bquad * 8;
  unsigned btile_off = lds_offset(btile) + (unsigned)(brow * 64 + bquad * 16);

  v8f a0 = {}, a1 = {}, a2 = {}, a3 = {};
  for (int kk = 0; kk < K; kk += 32) {
    // stage B tile for this K step: 256 x 16B async transfers -> 4KB in LDS
    async_load_b128(btile_off, bsrc + kk);
    __builtin_prefetch(arow + kk + 128, 0, 0);   // global_prefetch for A stream
    wait_asynccnt0();
    __syncthreads();

    v16h af  = frag_a(arow + kk, kbA);
    v16h bf0 = frag_b(&btile[(0  + n) * 32 + kbB]);
    v16h bf1 = frag_b(&btile[(16 + n) * 32 + kbB]);
    v16h bf2 = frag_b(&btile[(32 + n) * 32 + kbB]);
    v16h bf3 = frag_b(&btile[(48 + n) * 32 + kbB]);
    a0 = wmma16(af, bf0, a0);
    a1 = wmma16(af, bf1, a1);
    a2 = wmma16(af, bf2, a2);
    a3 = wmma16(af, bf3, a3);
    __syncthreads();   // protect btile from next iteration's overwrite
  }

  int mb = (lane < 16) ? 0 : 8;
#pragma unroll
  for (int t = 0; t < 4; ++t) {
    v8f acc = (t == 0) ? a0 : (t == 1) ? a1 : (t == 2) ? a2 : a3;
    int col = col0 + 16 * t + n;
    float bv = bias[col];
#pragma unroll
    for (int i = 0; i < 8; ++i) {
      int row = row0 + mb + i;
      float v = acc[i] + bv;
      if (act == 1) v = gelu_exact(v);
      if (res)  v += res[(long)row * N + col];
      if (outF) outF[(long)row * N + col] = v;
      if (outH) {
        if (transH) outH[(long)col * M + row] = (_Float16)v;
        else        outH[(long)row * N + col] = (_Float16)v;
      }
    }
  }
}

// ---------------------------------------------------------------------------
// Attention pass 1: per-(h,b), per key column k compute column-softmax stats
// over the QUERY axis: m_k = max_q s(q,k), Z_k = sum_q exp(s(q,k)-m_k).
// One wave owns 16 key columns; sweeps all S query rows with WMMA score tiles.
// ---------------------------------------------------------------------------
__global__ void attn_stats(const _Float16* __restrict__ q,
                           const _Float16* __restrict__ k,
                           float* __restrict__ mst, float* __restrict__ zst) {
  int hb = blockIdx.y;
  int h = hb / B_, b = hb % B_;
  int lane = threadIdx.x & 31, wave = threadIdx.x >> 5;
  int k0 = (blockIdx.x * 8 + wave) * 16;

  const _Float16* qb = q + ((long)h * M_ + (long)b * S_) * HD_;
  const _Float16* kb = k + ((long)h * M_ + (long)b * S_) * HD_;

  int n   = lane & 15;
  int kbA = (lane < 16) ? 0 : 16;
  int kbB = (lane < 16) ? 0 : 16;

  // K-side B fragments are fixed for the whole query sweep (HD=64 -> 2 steps)
  const _Float16* krow = kb + (long)(k0 + n) * HD_;
  v16h bf0 = frag_b(krow + 0  + kbB);
  v16h bf1 = frag_b(krow + 32 + kbB);

  float mrun = -3.0e38f, zrun = 0.f;
  for (int q0 = 0; q0 < S_; q0 += 16) {
    const _Float16* qrow = qb + (long)(q0 + n) * HD_;  // lane m == lane&15
    v16h a0 = frag_a(qrow + 0,  kbA);
    v16h a1 = frag_a(qrow + 32, kbA);
    v8f s = {};
    s = wmma16(a0, bf0, s);
    s = wmma16(a1, bf1, s);
    float tmax = -3.0e38f;
#pragma unroll
    for (int i = 0; i < 8; ++i) { s[i] *= SCALE_; tmax = fmaxf(tmax, s[i]); }
    tmax = fmaxf(tmax, __shfl_xor(tmax, 16, 32));   // combine lane pair (rows 0-7 / 8-15)
    float tsum = 0.f;
#pragma unroll
    for (int i = 0; i < 8; ++i) tsum += __expf(s[i] - tmax);
    tsum += __shfl_xor(tsum, 16, 32);
    float nm = fmaxf(mrun, tmax);
    zrun = zrun * __expf(mrun - nm) + tsum * __expf(tmax - nm);
    mrun = nm;
  }
  if (lane < 16) {
    mst[(long)hb * S_ + k0 + n] = mrun;
    zst[(long)hb * S_ + k0 + n] = zrun;
  }
}

// ---------------------------------------------------------------------------
// Attention pass 2: per-(h,b), per 16-row query tile: recompute score tiles,
// p = exp(s - m_k)/Z_k, re-layout p via LDS (C-frag -> A-frag), accumulate
// O[16 x 64] += P(16x32) x V(32x64) with WMMA. V is pre-transposed [HD][M].
// ---------------------------------------------------------------------------
__global__ void attn_apply(const _Float16* __restrict__ q,
                           const _Float16* __restrict__ k,
                           const _Float16* __restrict__ vT,
                           const float* __restrict__ mst,
                           const float* __restrict__ zst,
                           _Float16* __restrict__ o) {
  __shared__ _Float16 plds[8][16 * 32];   // per-wave 16(q) x 32(k) P tile
  int hb = blockIdx.y;
  int h = hb / B_, b = hb % B_;
  int lane = threadIdx.x & 31, wave = threadIdx.x >> 5;
  int q0 = (blockIdx.x * 8 + wave) * 16;

  const _Float16* qb = q + ((long)h * M_ + (long)b * S_) * HD_;
  const _Float16* kb = k + ((long)h * M_ + (long)b * S_) * HD_;
  const _Float16* vb = vT + (long)h * HD_ * M_ + (long)b * S_;  // row e: vb + e*M_ + s

  int n   = lane & 15;
  int kbA = (lane < 16) ? 0 : 16;
  int kbB = (lane < 16) ? 0 : 16;
  int mb  = (lane < 16) ? 0 : 8;

  // Q A-fragments fixed for whole key sweep
  const _Float16* qrow = qb + (long)(q0 + n) * HD_;
  v16h qa0 = frag_a(qrow + 0,  kbA);
  v16h qa1 = frag_a(qrow + 32, kbA);

  v8f o0 = {}, o1 = {}, o2 = {}, o3 = {};
  _Float16* pl = &plds[wave][0];

  for (int kk0 = 0; kk0 < S_; kk0 += 32) {
#pragma unroll
    for (int t = 0; t < 2; ++t) {                 // two 16-col score tiles
      int kt = kk0 + t * 16;
      const _Float16* krow = kb + (long)(kt + n) * HD_;
      v16h bf0 = frag_b(krow + 0  + kbB);
      v16h bf1 = frag_b(krow + 32 + kbB);
      v8f s = {};
      s = wmma16(qa0, bf0, s);
      s = wmma16(qa1, bf1, s);
      float mc = mst[(long)hb * S_ + kt + n];
      float rz = 1.0f / zst[(long)hb * S_ + kt + n];
#pragma unroll
      for (int i = 0; i < 8; ++i) {
        float p = __expf(s[i] * SCALE_ - mc) * rz;
        pl[(mb + i) * 32 + t * 16 + n] = (_Float16)p;   // C-layout -> row-major LDS
      }
    }
    // re-read P in A-fragment layout (wave-private LDS; DS ops are in-order)
    const _Float16* prow = pl + n * 32;                 // lane m == lane&15
    v16h pa = frag_a(prow, kbA);
    // V B-fragments: lane n -> output feature column e = 16t+n, contiguous in key index
#pragma unroll
    for (int t = 0; t < 4; ++t) {
      v16h vf = frag_b(vb + (long)(t * 16 + n) * M_ + kk0 + kbB);
      v8f acc = (t == 0) ? o0 : (t == 1) ? o1 : (t == 2) ? o2 : o3;
      acc = wmma16(pa, vf, acc);
      if (t == 0) o0 = acc; else if (t == 1) o1 = acc;
      else if (t == 2) o2 = acc; else o3 = acc;
    }
  }

  // write O tile, concatenating heads on the feature axis
#pragma unroll
  for (int t = 0; t < 4; ++t) {
    v8f acc = (t == 0) ? o0 : (t == 1) ? o1 : (t == 2) ? o2 : o3;
    int col = h * HD_ + t * 16 + n;
#pragma unroll
    for (int i = 0; i < 8; ++i) {
      long row = (long)b * S_ + q0 + mb + i;
      o[row * D_ + col] = (_Float16)acc[i];
    }
  }
}

// ---------------------------------------------------------------------------
// Host-side orchestration
// ---------------------------------------------------------------------------
extern "C" void kernel_launch(void* const* d_in, const int* in_sizes, int n_in,
                              void* d_out, int out_size, void* d_ws, size_t ws_size,
                              hipStream_t stream) {
  (void)in_sizes; (void)n_in; (void)out_size; (void)ws_size;
  const float* x    = (const float*)d_in[0];
  const float* ln1g = (const float*)d_in[1];
  const float* ln1b = (const float*)d_in[2];
  const float* ln2g = (const float*)d_in[3];
  const float* ln2b = (const float*)d_in[4];
  const float* Wq   = (const float*)d_in[5];
  const float* bq   = (const float*)d_in[6];
  const float* Wk   = (const float*)d_in[7];
  const float* bk   = (const float*)d_in[8];
  const float* Wv   = (const float*)d_in[9];
  const float* bv   = (const float*)d_in[10];
  const float* Wo   = (const float*)d_in[11];
  const float* bo   = (const float*)d_in[12];
  const float* W1   = (const float*)d_in[13];
  const float* b1   = (const float*)d_in[14];
  const float* W2   = (const float*)d_in[15];
  const float* b2   = (const float*)d_in[16];

  char* base = (char*)d_ws;
  size_t off = 0;
  auto carve = [&](size_t bytes) -> void* {
    void* p = base + off;
    off += (bytes + 255) & ~(size_t)255;
    return p;
  };
  const size_t HF = sizeof(_Float16), FF = sizeof(float);
  _Float16* Wq_h  = (_Float16*)carve(HF * (size_t)H_ * HD_ * D_);
  _Float16* Wk_h  = (_Float16*)carve(HF * (size_t)H_ * HD_ * D_);
  _Float16* Wv_h  = (_Float16*)carve(HF * (size_t)H_ * HD_ * D_);
  _Float16* Wo_h  = (_Float16*)carve(HF * (size_t)D_ * D_);
  _Float16* W1_h  = (_Float16*)carve(HF * (size_t)D_ * I_);
  _Float16* W2_h  = (_Float16*)carve(HF * (size_t)I_ * D_);
  _Float16* h_h   = (_Float16*)carve(HF * (size_t)M_ * D_);
  _Float16* q_h   = (_Float16*)carve(HF * (size_t)H_ * M_ * HD_);
  _Float16* k_h   = (_Float16*)carve(HF * (size_t)H_ * M_ * HD_);
  _Float16* vT_h  = (_Float16*)carve(HF * (size_t)H_ * HD_ * M_);
  float*    mst   = (float*)   carve(FF * (size_t)H_ * B_ * S_);
  float*    zst   = (float*)   carve(FF * (size_t)H_ * B_ * S_);
  _Float16* o_h   = (_Float16*)carve(HF * (size_t)M_ * D_);
  float*    x1    = (float*)   carve(FF * (size_t)M_ * D_);
  float*    xn    = (float*)   carve(FF * (size_t)M_ * D_);
  _Float16* xn_h  = (_Float16*)carve(HF * (size_t)M_ * D_);
  _Float16* ff1_h = (_Float16*)carve(HF * (size_t)M_ * I_);
  // total scratch ~103 MB

  dim3 blk(256);

  // 1) weights -> transposed f16
  cvt_transpose<<<dim3(256, H_), blk, 0, stream>>>(Wq, Wq_h, D_, HD_);
  cvt_transpose<<<dim3(256, H_), blk, 0, stream>>>(Wk, Wk_h, D_, HD_);
  cvt_transpose<<<dim3(256, H_), blk, 0, stream>>>(Wv, Wv_h, D_, HD_);
  cvt_transpose<<<dim3(256, 1),  blk, 0, stream>>>(Wo, Wo_h, D_, D_);
  cvt_transpose<<<dim3(256, 1),  blk, 0, stream>>>(W1, W1_h, D_, I_);
  cvt_transpose<<<dim3(256, 1),  blk, 0, stream>>>(W2, W2_h, I_, D_);

  // 2) LN1 -> h (f16)
  ln_kernel<<<dim3(M_), blk, 0, stream>>>(x, ln1g, ln1b, nullptr, h_h, D_);

  // 3) Q, K, V projections (batched over heads); V written transposed [HD][M]
  gemm_wmma<<<dim3(1, M_ / 128, H_), blk, 0, stream>>>(
      h_h, Wq_h, bq, nullptr, nullptr, q_h, M_, HD_, D_,
      (long)HD_ * D_, (long)HD_, (long)M_ * HD_, 0, 0);
  gemm_wmma<<<dim3(1, M_ / 128, H_), blk, 0, stream>>>(
      h_h, Wk_h, bk, nullptr, nullptr, k_h, M_, HD_, D_,
      (long)HD_ * D_, (long)HD_, (long)M_ * HD_, 0, 0);
  gemm_wmma<<<dim3(1, M_ / 128, H_), blk, 0, stream>>>(
      h_h, Wv_h, bv, nullptr, nullptr, vT_h, M_, HD_, D_,
      (long)HD_ * D_, (long)HD_, (long)M_ * HD_, 0, 1);

  // 4) column-softmax stats (softmax over QUERY axis), then apply + PV
  attn_stats<<<dim3(S_ / 128, H_ * B_), blk, 0, stream>>>(q_h, k_h, mst, zst);
  attn_apply<<<dim3(S_ / 128, H_ * B_), blk, 0, stream>>>(q_h, k_h, vT_h, mst, zst, o_h);

  // 5) output projection + residual with x -> x1 (f32)
  gemm_wmma<<<dim3(D_ / 64, M_ / 128, 1), blk, 0, stream>>>(
      o_h, Wo_h, bo, x, x1, nullptr, M_, D_, D_, 0, 0, 0, 0, 0);

  // 6) LN2 -> xn (f32 for residual) + xn_h (f16 for GEMM)
  ln_kernel<<<dim3(M_), blk, 0, stream>>>(x1, ln2g, ln2b, xn, xn_h, D_);

  // 7) FFN: gelu(xn @ W1 + b1) -> ff1 (f16); ff1 @ W2 + b2 + xn -> d_out (f32)
  gemm_wmma<<<dim3(I_ / 64, M_ / 128, 1), blk, 0, stream>>>(
      xn_h, W1_h, b1, nullptr, nullptr, ff1_h, M_, I_, D_, 0, 0, 0, 1, 0);
  gemm_wmma<<<dim3(D_ / 64, M_ / 128, 1), blk, 0, stream>>>(
      ff1_h, W2_h, b2, xn, (float*)d_out, nullptr, M_, D_, I_, 0, 0, 0, 0, 0);
}
